// NLPModel_39006892982823
// MI455X (gfx1250) — matrix operand order repaired and checked by reading, not compiled, not run on hardware
//
#include <hip/hip_runtime.h>

// ---------------------------------------------------------------------------
// minGRU LM forward for MI455X (gfx1250, wave32, WMMA).
// Dominant cost: logits GEMM [8192,512]x[512,32000] (268 GFLOP) + 1.05 GB
// f32 output store (~45us @ 23.3 TB/s). bf16 WMMA (16x16x32) makes compute
// comparable to the mandatory store; A (8.4MB) and B^T (32MB) live in L2.
// GEMM waves compute 32x64 tiles (8 accumulators) so each B-fragment load is
// reused by two WMMAs: 12 b128 loads per 8 WMMAs.
// ---------------------------------------------------------------------------

typedef __bf16 bf16_t;
typedef bf16_t v16bf __attribute__((ext_vector_type(16)));
typedef float  v8f   __attribute__((ext_vector_type(8)));
typedef unsigned short ushort8 __attribute__((ext_vector_type(8)));

#define BB    4
#define SS    2048
#define EMB   512
#define HID   512
#define VOCAB 32000
#define MROWS (BB * SS)   /* 8192 */

static __device__ __forceinline__ bf16_t f2bf(float f) {
    union { float f; unsigned int u; } in; in.f = f;
    unsigned int u = in.u;
    u += 0x7FFFu + ((u >> 16) & 1u);          // round-to-nearest-even
    union { unsigned short s; bf16_t b; } out; out.s = (unsigned short)(u >> 16);
    return out.b;
}

static __device__ __forceinline__ float softplus_f(float v) {
    // log(1 + e^v), numerically stable
    return fmaxf(v, 0.f) + log1pf(expf(-fabsf(v)));
}

// --------------------------- embedding gather ------------------------------
__global__ void __launch_bounds__(256)
k_gather_emb(const int* __restrict__ ids, const float* __restrict__ emb,
             float* __restrict__ x) {
    const int D4 = EMB / 4;                       // 128 float4 per row
    int idx = blockIdx.x * blockDim.x + threadIdx.x;
    if (idx >= MROWS * D4) return;
    int row = idx / D4, c = idx - row * D4;
    ((float4*)x)[idx] = ((const float4*)emb)[(size_t)ids[row] * D4 + c];
}

// ------------------- f32 [K,N] -> bf16 [N,K] transpose ---------------------
__global__ void __launch_bounds__(256)
k_transpose_bf16(const float* __restrict__ in, bf16_t* __restrict__ out,
                 int K, int N) {
    __shared__ float tile[32][33];
    int nx = blockIdx.x * 32 + threadIdx.x;       // N
    int ky = blockIdx.y * 32 + threadIdx.y;       // K (y in 0..7)
#pragma unroll
    for (int j = 0; j < 32; j += 8)
        tile[threadIdx.y + j][threadIdx.x] = in[(size_t)(ky + j) * N + nx];
    __syncthreads();
    int n = blockIdx.x * 32 + threadIdx.y;
    int k = blockIdx.y * 32 + threadIdx.x;
#pragma unroll
    for (int j = 0; j < 32; j += 8)
        out[(size_t)(n + j) * K + k] = f2bf(tile[threadIdx.x][threadIdx.y + j]);
}

// ------------------- wave-per-row LayerNorm -> bf16 ------------------------
// D = 512, one wave32 per row, 16 elements per lane.
__global__ void __launch_bounds__(256)
k_layernorm_bf16(const float* __restrict__ x, const float* __restrict__ g,
                 const float* __restrict__ b, bf16_t* __restrict__ out,
                 int rows, float eps) {
    const int D = 512;
    int wave = threadIdx.x >> 5;
    int lane = threadIdx.x & 31;
    int row  = blockIdx.x * 8 + wave;
    if (row >= rows) return;
    const float* xr = x + (size_t)row * D;

    float4 v[4];
    float sum = 0.f;
#pragma unroll
    for (int j = 0; j < 4; ++j) {
        v[j] = ((const float4*)xr)[lane * 4 + j];
        sum += v[j].x + v[j].y + v[j].z + v[j].w;
    }
#pragma unroll
    for (int o = 16; o > 0; o >>= 1) sum += __shfl_xor(sum, o, 32);
    float mu = sum * (1.f / D);

    float vs = 0.f;
#pragma unroll
    for (int j = 0; j < 4; ++j) {
        float dx = v[j].x - mu, dy = v[j].y - mu, dz = v[j].z - mu, dw = v[j].w - mu;
        vs += dx * dx + dy * dy + dz * dz + dw * dw;
    }
#pragma unroll
    for (int o = 16; o > 0; o >>= 1) vs += __shfl_xor(vs, o, 32);
    float inv = rsqrtf(vs * (1.f / D) + eps);

    bf16_t* orow = out + (size_t)row * D;
#pragma unroll
    for (int j = 0; j < 4; ++j) {
        int d = lane * 16 + j * 4;
        float e0 = (v[j].x - mu) * inv * g[d + 0] + b[d + 0];
        float e1 = (v[j].y - mu) * inv * g[d + 1] + b[d + 1];
        float e2 = (v[j].z - mu) * inv * g[d + 2] + b[d + 2];
        float e3 = (v[j].w - mu) * inv * g[d + 3] + b[d + 3];
        orow[d + 0] = f2bf(e0);
        orow[d + 1] = f2bf(e1);
        orow[d + 2] = f2bf(e2);
        orow[d + 3] = f2bf(e3);
    }
}

// --------------------------- WMMA bf16 GEMM --------------------------------
// C[M,N] = A[M,K] * B[K,N] (+bias), A row-major bf16, Bt = B^T [N,K] bf16.
// Block: 256 threads = 8 waves. Wave w -> rows m0 = by*256 + w*32 (two
// 16-row tiles), cols n0 = bx*64 (four 16-col tiles) -> 8 accumulators.
// Each B-fragment load feeds both M-tiles (register-level reuse; operands
// are L2-resident so no LDS staging needed).
// Fragment addressing (A and B share a contiguous K order, not the ISA table order):
//   A 16x32: lane&15 = M row; lane>>4 selects K half 0..15 vs 16..31
//   B 32x16: lane&15 = N col; lane>>4 selects K half 0..15 vs 16..31
//   C/D:     lane&15 = N col; lane>>4 adds 8 to M; VGPR r = M row offset
__global__ void __launch_bounds__(256)
k_wmma_gemm_bf16(const bf16_t* __restrict__ A, const bf16_t* __restrict__ Bt,
                 const float* __restrict__ bias, float* __restrict__ C,
                 int M, int N, int K) {
    const int lane = threadIdx.x & 31;
    const int wave = threadIdx.x >> 5;
    const int m0   = blockIdx.y * 256 + wave * 32;
    const int n0   = blockIdx.x * 64;
    const int half = lane >> 4;      // 0 or 1
    const int l16  = lane & 15;
    if (m0 >= M) return;

    v8f acc[2][4] = {};
    const bf16_t* arow0 = A + (size_t)(m0 + l16) * K;        // M-tile 0
    const bf16_t* arow1 = arow0 + (size_t)16 * K;            // M-tile 1

    for (int k0 = 0; k0 < K; k0 += 32) {
        union { ushort8 u[2]; v16bf v; } af0, af1;
        af0.u[0] = *(const ushort8*)(arow0 + k0 + half * 16);      // K +0..7 / +16..23
        af0.u[1] = *(const ushort8*)(arow0 + k0 + half * 16 + 8);  // K +8..15 / +24..31
        af1.u[0] = *(const ushort8*)(arow1 + k0 + half * 16);
        af1.u[1] = *(const ushort8*)(arow1 + k0 + half * 16 + 8);
#pragma unroll
        for (int t = 0; t < 4; ++t) {
            const bf16_t* bp = Bt + (size_t)(n0 + t * 16 + l16) * K + k0 + half * 16;
            v16bf bf = *(const v16bf*)bp;                    // 16 contiguous K values
            acc[0][t] = __builtin_amdgcn_wmma_f32_16x16x32_bf16(
                false, af0.v, false, bf, (short)0, acc[0][t], false, false);
            acc[1][t] = __builtin_amdgcn_wmma_f32_16x16x32_bf16(
                false, af1.v, false, bf, (short)0, acc[1][t], false, false);
        }
    }

#pragma unroll
    for (int mt = 0; mt < 2; ++mt) {
#pragma unroll
        for (int t = 0; t < 4; ++t) {
            int n = n0 + t * 16 + l16;
            float bv = bias ? bias[n] : 0.f;
#pragma unroll
            for (int r = 0; r < 8; ++r) {
                int m = m0 + mt * 16 + r + half * 8;
                C[(size_t)m * N + n] = acc[mt][t][r] + bv;
            }
        }
    }
}

// ------------------------- minGRU log-domain scan --------------------------
// pre: [B,S,2H] (k | h_pre). x: [B,S,H], updated in place: x += exp(log_h).
// One thread per (b,h) channel; online running-(max,sum) logcumsumexp.
__global__ void __launch_bounds__(256)
k_mingru_scan(const float* __restrict__ pre, float* __restrict__ x) {
    int tid = blockIdx.x * blockDim.x + threadIdx.x;
    if (tid >= BB * HID) return;
    int b = tid / HID, h = tid - b * HID;
    const float* pk = pre + (size_t)b * SS * 2 * HID + h;
    const float* ph = pk + HID;
    float* px = x + (size_t)b * SS * HID + h;

    float a_star = 0.f;
    float m = -__builtin_inff();
    float ssum = 0.f;
    for (int s = 0; s < SS; ++s) {
        float kk = pk[(size_t)s * 2 * HID];
        float hp = ph[(size_t)s * 2 * HID];
        float log_coeff = -softplus_f(kk);                 // log(1 - z)
        float log_g = (hp >= 0.f) ? logf(hp + 0.5f) : -softplus_f(-hp);
        float log_b = -softplus_f(-kk) + log_g;            // log z + log g(h~)
        a_star += log_coeff;
        float v = log_b - a_star;
        float mn = fmaxf(m, v);
        ssum = ssum * expf(m - mn) + expf(v - mn);
        m = mn;
        float log_h = a_star + m + logf(ssum);
        px[(size_t)s * HID] += expf(log_h);                // h + residual
    }
}

// ---------------------------------------------------------------------------
extern "C" void kernel_launch(void* const* d_in, const int* in_sizes, int n_in,
                              void* d_out, int out_size, void* d_ws, size_t ws_size,
                              hipStream_t stream) {
    (void)in_sizes; (void)n_in; (void)out_size; (void)ws_size;
    const int*   ids   = (const int*)  d_in[0];
    const float* emb   = (const float*)d_in[1];
    const float* W0    = (const float*)d_in[2];
    const float* W1    = (const float*)d_in[3];
    const float* ln0_g = (const float*)d_in[4];
    const float* ln0_b = (const float*)d_in[5];
    const float* ln1_g = (const float*)d_in[6];
    const float* ln1_b = (const float*)d_in[7];
    const float* lnf_g = (const float*)d_in[8];
    const float* lnf_b = (const float*)d_in[9];
    const float* fc_w  = (const float*)d_in[10];
    const float* fc_b  = (const float*)d_in[11];
    float* out = (float*)d_out;

    // workspace layout (all sizes 256B-aligned)
    char* ws = (char*)d_ws;
    float*  x    = (float*)ws;   ws += (size_t)MROWS * EMB * 4;        // 16 MB
    bf16_t* xn   = (bf16_t*)ws;  ws += (size_t)MROWS * EMB * 2;        //  8 MB
    float*  pre  = (float*)ws;   ws += (size_t)MROWS * 2 * HID * 4;    // 33 MB
    bf16_t* w0t  = (bf16_t*)ws;  ws += (size_t)(2 * HID) * EMB * 2;    //  1 MB
    bf16_t* w1t  = (bf16_t*)ws;  ws += (size_t)(2 * HID) * HID * 2;    //  1 MB
    bf16_t* fct  = (bf16_t*)ws;  ws += (size_t)VOCAB * HID * 2;        // 32 MB

    dim3 tb(32, 8);
    k_transpose_bf16<<<dim3((2 * HID) / 32, EMB / 32), tb, 0, stream>>>(W0, w0t, EMB, 2 * HID);
    k_transpose_bf16<<<dim3((2 * HID) / 32, HID / 32), tb, 0, stream>>>(W1, w1t, HID, 2 * HID);
    k_transpose_bf16<<<dim3(VOCAB / 32, HID / 32),     tb, 0, stream>>>(fc_w, fct, HID, VOCAB);

    k_gather_emb<<<(MROWS * (EMB / 4) + 255) / 256, 256, 0, stream>>>(ids, emb, x);

    // layer 0
    k_layernorm_bf16<<<MROWS / 8, 256, 0, stream>>>(x, ln0_g, ln0_b, xn, MROWS, 1e-5f);
    k_wmma_gemm_bf16<<<dim3((2 * HID) / 64, MROWS / 256), 256, 0, stream>>>(
        xn, w0t, nullptr, pre, MROWS, 2 * HID, EMB);
    k_mingru_scan<<<(BB * HID) / 256, 256, 0, stream>>>(pre, x);

    // layer 1
    k_layernorm_bf16<<<MROWS / 8, 256, 0, stream>>>(x, ln1_g, ln1_b, xn, MROWS, 1e-5f);
    k_wmma_gemm_bf16<<<dim3((2 * HID) / 64, MROWS / 256), 256, 0, stream>>>(
        xn, w1t, nullptr, pre, MROWS, 2 * HID, HID);
    k_mingru_scan<<<(BB * HID) / 256, 256, 0, stream>>>(pre, x);

    // final norm + logits
    k_layernorm_bf16<<<MROWS / 8, 256, 0, stream>>>(x, lnf_g, lnf_b, xn, MROWS, 0.0f);
    k_wmma_gemm_bf16<<<dim3(VOCAB / 64, MROWS / 256), 256, 0, stream>>>(
        xn, fct, fc_b, out, MROWS, VOCAB, HID);
}
